// MultiHead_Group_Query_Attention_19739669692948
// MI455X (gfx1250) — compile-verified, matrix operand order checked by reading, not run on hardware
//
#include <hip/hip_runtime.h>
#include <hip/hip_bf16.h>
#include <math.h>

// ---------------- problem constants ----------------
static constexpr int Bq   = 2;
static constexpr int S    = 2048;
static constexpr int D    = 2048;
static constexpr int NQ   = 16;
static constexpr int NKV  = 4;
static constexpr int HD   = 128;          // D / NQ
static constexpr int GRP  = NQ / NKV;     // 4
static constexpr int M    = Bq * S;       // 4096 rows of x
static constexpr float QSCALE = 0.08838834764831845f; // 1/sqrt(128)

// ---------------- CDNA5 types ----------------
typedef __bf16 bf16;
typedef __attribute__((ext_vector_type(16))) __bf16 v16bf;
typedef __attribute__((ext_vector_type(8)))  __bf16 v8bf;
typedef __attribute__((ext_vector_type(8)))  float  v8f;
typedef __attribute__((ext_vector_type(4)))  unsigned int v4u;
typedef __attribute__((ext_vector_type(8)))  int  v8i32;
typedef __attribute__((ext_vector_type(4)))  int  v4i32;

static __device__ inline v16bf combine16(v8bf lo, v8bf hi) {
    v16bf r;
#pragma unroll
    for (int i = 0; i < 8; ++i) { r[i] = lo[i]; r[8 + i] = hi[i]; }
    return r;
}

// Load a 16x32 bf16 fragment (A layout, ISA 7.12.2) from a row-major source.
// lane&15 selects the row; lanes>=16 take the K+8 halves.
static __device__ inline v16bf load_frag_g(const bf16* base, int stride, int lane) {
    const int r  = lane & 15;
    const int kb = (lane >> 4) * 8;
    const bf16* p = base + (size_t)r * stride + kb;
    v8bf lo = *(const v8bf*)p;
    v8bf hi = *(const v8bf*)(p + 16);
    return combine16(lo, hi);
}

// ---------------------------------------------------------------------
// Tensor Data Mover: DMA a 2-D bf16 tile (tile_d0 x tile_d1, row-major,
// row stride stride0 elems) from global into LDS at byte offset lds_off.
// pad: insert 16B into LDS after every 64B stored (ISA 8.4 group1
// pad_interval=3 / pad_amount=3) -> LDS row stride 80B = 40 bf16.
// ---------------------------------------------------------------------
static __device__ inline void tdm_load_tile_2d_bf16(
    unsigned lds_off, const void* gptr,
    unsigned tensor_d0, unsigned tensor_d1,
    unsigned long long stride0, unsigned tile_d0, unsigned tile_d1) {
    const unsigned long long ga = (unsigned long long)gptr;
    v4u g0;
    g0[0] = 1u;                                        // count=1, user D#
    g0[1] = lds_off;                                   // lds_addr
    g0[2] = (unsigned)(ga & 0xFFFFFFFFu);              // global_addr[31:0]
    g0[3] = (unsigned)((ga >> 32) & 0x01FFFFFFu)       // global_addr[56:32]
          | (2u << 30);                                // type = 2 (image)
    v8i32 g1;
    g1[0] = (int)((1u << 16)                           // data_size = 2B
          | (1u << 20)                                 // pad_enable
          | (3u << 22)                                 // pad_interval: 16 DW (64B)
          | (3u << 25));                               // pad_amount: 4 DW (16B)
    g1[1] = (int)((tensor_d0 & 0xFFFFu) << 16);                          // t_dim0 lo
    g1[2] = (int)((tensor_d0 >> 16) | ((tensor_d1 & 0xFFFFu) << 16));    // t_dim0 hi | t_dim1 lo
    g1[3] = (int)((tensor_d1 >> 16) | (tile_d0 << 16));                  // t_dim1 hi | tile_dim0
    g1[4] = (int)(tile_d1 & 0xFFFFu);                                    // tile_dim1 (tile_dim2=0)
    g1[5] = (int)(unsigned)(stride0 & 0xFFFFFFFFu);                      // dim0_stride lo
    g1[6] = (int)(unsigned)((stride0 >> 32) & 0xFFFFu);                  // dim0_stride hi
    g1[7] = 0;
    v4i32 gz = {};
#if defined(__clang_major__) && (__clang_major__ >= 23)
    v8i32 gz8 = {};
    __builtin_amdgcn_tensor_load_to_lds(g0, g1, gz, gz, gz8, 0);
#else
    __builtin_amdgcn_tensor_load_to_lds(g0, g1, gz, gz, 0);
#endif
}

// =====================================================================
// Generic GEMM: C[M x N] = A[M x K] * B[N x K]^T  (torch Linear layout)
// Block tile 128x64, BK=32, 256 threads = 8 waves (2 M x 4 N).
// A path: fp32 (VALU-converted during LDS fill)  OR  bf16 via TDM DMA.
// B (weights) is fp32, converted during fill.
// mode 0: fp32 row-major out; mode 1: bf16 head-split [b,h,s,hd] *scale;
// mode 2: bf16 head-split transposed [b,h,hd,s] *scale.
// =====================================================================
__global__ __launch_bounds__(256)
void gemm_bt_kernel(const float* __restrict__ A32,
                    const bf16*  __restrict__ A16, int a_is_bf16,
                    const float* __restrict__ Bw,
                    int N, int K,
                    float* __restrict__ out32, bf16* __restrict__ outbf,
                    int mode, int H, float scale) {
    __shared__ bf16 As[128 * 40];   // 80B row stride == TDM pad layout
    __shared__ bf16 Bs[64 * 40];

    const int tid   = threadIdx.x;
    const int tileM = blockIdx.y * 128;
    const int tileN = blockIdx.x * 64;

    const int lane = tid & 31;
    const int w    = tid >> 5;
    const int wm   = (w & 1) * 64;   // wave M offset (4 subtiles of 16)
    const int wn   = (w >> 1) * 16;  // wave N offset
    const int r16  = lane & 15;
    const int kb   = (lane >> 4) * 8;

    // LDS byte offset of As: flat LDS addresses carry offset in addr[31:0]
    const unsigned as_base = (unsigned)(size_t)(void*)&As[0];

    v8f acc[4] = {v8f{}, v8f{}, v8f{}, v8f{}};

    for (int k0 = 0; k0 < K; k0 += 32) {
        __syncthreads();   // previous iteration finished reading LDS
        // ---- fill A tile: 128 rows x 32 cols ----
        if (a_is_bf16) {
            if (w == 0)    // one TDM DMA per block (EXEC-independent, TENSORcnt)
                tdm_load_tile_2d_bf16(as_base,
                                      A16 + (size_t)tileM * K + k0,
                                      (unsigned)K, (unsigned)M,
                                      (unsigned long long)K, 32u, 128u);
        } else {
            const int r  = tid >> 1;
            const int cb = (tid & 1) * 16;
            const float4* s4 = (const float4*)(A32 + (size_t)(tileM + r) * K + k0 + cb);
#pragma unroll
            for (int q = 0; q < 2; ++q) {
                float4 f0 = s4[2 * q], f1 = s4[2 * q + 1];
                v8bf o;
                o[0] = (bf16)f0.x; o[1] = (bf16)f0.y; o[2] = (bf16)f0.z; o[3] = (bf16)f0.w;
                o[4] = (bf16)f1.x; o[5] = (bf16)f1.y; o[6] = (bf16)f1.z; o[7] = (bf16)f1.w;
                *(v8bf*)&As[r * 40 + cb + 8 * q] = o;
            }
        }
        // ---- fill B tile: 64 rows (of W) x 32 cols, 8 elems/thread ----
        {
            const int r  = tid >> 2;
            const int cb = (tid & 3) * 8;
            const float4* s4 = (const float4*)(Bw + (size_t)(tileN + r) * K + k0 + cb);
            float4 f0 = s4[0], f1 = s4[1];
            v8bf o;
            o[0] = (bf16)f0.x; o[1] = (bf16)f0.y; o[2] = (bf16)f0.z; o[3] = (bf16)f0.w;
            o[4] = (bf16)f1.x; o[5] = (bf16)f1.y; o[6] = (bf16)f1.z; o[7] = (bf16)f1.w;
            *(v8bf*)&Bs[r * 40 + cb] = o;
        }
        // prefetch next K tile into L2 (global_prefetch_b8)
        if (k0 + 32 < K) {
            __builtin_prefetch(Bw + (size_t)(tileN + (tid >> 2)) * K + k0 + 32 + (tid & 3) * 8, 0, 1);
            if (!a_is_bf16)
                __builtin_prefetch(A32 + (size_t)(tileM + (tid >> 1)) * K + k0 + 32 + (tid & 1) * 16, 0, 1);
        }
        if (a_is_bf16 && w == 0)
            __builtin_amdgcn_s_wait_tensorcnt(0);   // DMA landed in LDS
        __syncthreads();

        // ---- B fragment (shared by the wave's 4 M-subtiles) ----
        v8bf blo = *(v8bf*)&Bs[(wn + r16) * 40 + kb];
        v8bf bhi = *(v8bf*)&Bs[(wn + r16) * 40 + kb + 16];
        v16bf bfrag = combine16(blo, bhi);

#pragma unroll
        for (int i = 0; i < 4; ++i) {
            v8bf alo = *(v8bf*)&As[(wm + i * 16 + r16) * 40 + kb];
            v8bf ahi = *(v8bf*)&As[(wm + i * 16 + r16) * 40 + kb + 16];
            v16bf af = combine16(alo, ahi);
            acc[i] = __builtin_amdgcn_wmma_f32_16x16x32_bf16(
                false, af, false, bfrag, (short)0, acc[i], false, false);
        }
    }

    // ---- epilogue ----
#pragma unroll
    for (int i = 0; i < 4; ++i) {
#pragma unroll
        for (int j = 0; j < 8; ++j) {
            const int m = tileM + wm + i * 16 + ((lane >> 4) << 3) + j;
            const int n = tileN + wn + r16;
            const float v = acc[i][j];
            if (mode == 0) {
                out32[(size_t)m * N + n] = v;
            } else {
                const int b = m >> 11, s = m & (S - 1);
                const int h = n >> 7, d = n & (HD - 1);
                const size_t idx = (mode == 1)
                    ? ((size_t)(b * H + h) * S + s) * HD + d
                    : ((size_t)(b * H + h) * HD + d) * S + s;
                outbf[idx] = (bf16)(v * scale);
            }
        }
    }
}

// =====================================================================
// Flash attention, causal GQA. 1 wave / (b, h, 16-query tile).
// 32-key chunks: 8 WMMAs QK^T + 8 WMMAs P*V per chunk; fp32 online softmax.
// Q pre-scaled by 1/sqrt(HD); V stored transposed [b,hk,HD,S].
// =====================================================================
__global__ __launch_bounds__(32)
void attn_kernel(const bf16* __restrict__ Qh, const bf16* __restrict__ Kh,
                 const bf16* __restrict__ Vt, bf16* __restrict__ Ctx) {
    __shared__ bf16 Pl[16 * 40];

    const int bid = blockIdx.x;
    const int qt  = bid & (S / 16 - 1);
    const int h   = (bid >> 7) & (NQ - 1);
    const int b   = bid >> 11;
    const int q0  = qt * 16;
    const int hk  = h / GRP;

    const int lane = threadIdx.x & 31;
    const int r16  = lane & 15;
    const int kb   = (lane >> 4) * 8;
    const int rj   = (lane >> 4) << 3;   // C-frag row base for this lane

    const bf16* Qbase = Qh + ((size_t)(b * NQ + h) * S) * HD;
    const bf16* Kbase = Kh + ((size_t)(b * NKV + hk) * S) * HD;
    const bf16* Vbase = Vt + ((size_t)(b * NKV + hk) * HD) * S;

    // Q fragments: 16 rows x 128 (4 chunks of K=32)
    v16bf afQ[4];
#pragma unroll
    for (int c = 0; c < 4; ++c)
        afQ[c] = load_frag_g(Qbase + (size_t)q0 * HD + 32 * c, HD, lane);

    float m_r[8], l_r[8];
    v8f accv[8];
#pragma unroll
    for (int j = 0; j < 8; ++j) { m_r[j] = -1e30f; l_r[j] = 0.f; }
#pragma unroll
    for (int f = 0; f < 8; ++f) accv[f] = v8f{};

    for (int k0 = 0; k0 < q0 + 16; k0 += 32) {
        // ---- scores: 16q x 32k, two 16x16 C fragments ----
        v8f sc0 = v8f{}, sc1 = v8f{};
#pragma unroll
        for (int c = 0; c < 4; ++c) {
            v16bf bk0 = load_frag_g(Kbase + (size_t)k0 * HD + 32 * c, HD, lane);
            v16bf bk1 = load_frag_g(Kbase + (size_t)(k0 + 16) * HD + 32 * c, HD, lane);
            sc0 = __builtin_amdgcn_wmma_f32_16x16x32_bf16(
                false, afQ[c], false, bk0, (short)0, sc0, false, false);
            sc1 = __builtin_amdgcn_wmma_f32_16x16x32_bf16(
                false, afQ[c], false, bk1, (short)0, sc1, false, false);
        }

        // ---- causal mask + online softmax (rows split lanes 0-15 / 16-31) ----
#pragma unroll
        for (int j = 0; j < 8; ++j) {
            const int qrow = q0 + rj + j;
            float s0 = (k0 + r16      > qrow) ? -1e30f : sc0[j];
            float s1 = (k0 + 16 + r16 > qrow) ? -1e30f : sc1[j];
            float cm = fmaxf(s0, s1);
#pragma unroll
            for (int off = 8; off > 0; off >>= 1)
                cm = fmaxf(cm, __shfl_xor(cm, off, 16));
            const float mn   = fmaxf(m_r[j], cm);
            const float corr = __expf(m_r[j] - mn);
            const float p0   = __expf(s0 - mn);
            const float p1   = __expf(s1 - mn);
            float rs = p0 + p1;
#pragma unroll
            for (int off = 8; off > 0; off >>= 1)
                rs += __shfl_xor(rs, off, 16);
            l_r[j] = l_r[j] * corr + rs;
            m_r[j] = mn;
#pragma unroll
            for (int f = 0; f < 8; ++f) accv[f][j] *= corr;
            Pl[(rj + j) * 40 + r16]      = (bf16)p0;
            Pl[(rj + j) * 40 + 16 + r16] = (bf16)p1;
        }
        __syncthreads();

        // ---- P (16x32) as A fragment ----
        v8bf plo = *(v8bf*)&Pl[r16 * 40 + kb];
        v8bf phi = *(v8bf*)&Pl[r16 * 40 + 16 + kb];
        v16bf pf = combine16(plo, phi);

        // ---- ctx += P * V : 8 d-subtiles, V^T rows are contiguous in k ----
#pragma unroll
        for (int f = 0; f < 8; ++f) {
            v16bf bv = load_frag_g(Vbase + (size_t)(f * 16) * S + k0, S, lane);
            accv[f] = __builtin_amdgcn_wmma_f32_16x16x32_bf16(
                false, pf, false, bv, (short)0, accv[f], false, false);
        }
        __syncthreads();
    }

    // ---- normalize + store ctx (bf16, [b, s, h*HD + d]) ----
#pragma unroll
    for (int f = 0; f < 8; ++f) {
#pragma unroll
        for (int j = 0; j < 8; ++j) {
            const int srow = q0 + rj + j;
            const int d    = f * 16 + r16;
            const float v  = accv[f][j] / l_r[j];
            Ctx[((size_t)(b * S + srow)) * D + h * HD + d] = (bf16)v;
        }
    }
}

// =====================================================================
extern "C" void kernel_launch(void* const* d_in, const int* in_sizes, int n_in,
                              void* d_out, int out_size, void* d_ws, size_t ws_size,
                              hipStream_t stream) {
    (void)in_sizes; (void)n_in; (void)out_size; (void)ws_size;
    const float* x  = (const float*)d_in[0];
    const float* Wq = (const float*)d_in[1];
    const float* Wk = (const float*)d_in[2];
    const float* Wv = (const float*)d_in[3];
    const float* Wo = (const float*)d_in[4];
    float* out = (float*)d_out;

    // workspace: Qh 16MB | Kh 4MB | Vt 4MB | Ctx 16MB  (all bf16, L2-resident)
    bf16* Qh  = (bf16*)d_ws;
    bf16* Kh  = Qh + (size_t)Bq * NQ * S * HD;
    bf16* Vt  = Kh + (size_t)Bq * NKV * S * HD;
    bf16* Ctx = Vt + (size_t)Bq * NKV * S * HD;

    const dim3 blk(256);
    // Q = x @ Wq^T  -> head-split, pre-scaled by 1/sqrt(HD)
    gemm_bt_kernel<<<dim3(D / 64, M / 128), blk, 0, stream>>>(
        x, nullptr, 0, Wq, D, D, nullptr, Qh, 1, NQ, QSCALE);
    // K = x @ Wk^T  -> head-split
    gemm_bt_kernel<<<dim3((NKV * HD) / 64, M / 128), blk, 0, stream>>>(
        x, nullptr, 0, Wk, NKV * HD, D, nullptr, Kh, 1, NKV, 1.0f);
    // V = x @ Wv^T  -> head-split TRANSPOSED [b,hk,HD,S]
    gemm_bt_kernel<<<dim3((NKV * HD) / 64, M / 128), blk, 0, stream>>>(
        x, nullptr, 0, Wv, NKV * HD, D, nullptr, Vt, 2, NKV, 1.0f);
    // attention
    attn_kernel<<<dim3(Bq * NQ * (S / 16)), dim3(32), 0, stream>>>(Qh, Kh, Vt, Ctx);
    // out = ctx @ Wo^T (fp32 out), A staged into LDS by the Tensor Data Mover
    gemm_bt_kernel<<<dim3(D / 64, M / 128), blk, 0, stream>>>(
        nullptr, Ctx, 1, Wo, D, D, out, nullptr, 0, 0, 1.0f);
}
